// Lion3DBackbone_46505905881636
// MI455X (gfx1250) — matrix operand order repaired and checked by reading, not compiled, not run on hardware
//
#include <hip/hip_runtime.h>
#include <hip/hip_bf16.h>
#include <stdint.h>

typedef __attribute__((ext_vector_type(16))) __bf16 v16bf;
typedef __attribute__((ext_vector_type(8)))  __bf16 v8bf;
typedef __attribute__((ext_vector_type(8)))  float  v8f;

#define C      128
#define GROUP  256
#define DIG    128      // 7-bit radix digits
#define RBITS  7

// window geometry (SPARSE_SHAPE=(32,1024,1024), WINDOW=(12,12,32))
#define WXc 12
#define WYc 12
#define WZc 32
#define MXc 87          // ceil(1024/12)+1
#define MYc 87
#define MZc 2           // ceil(32/32)+1
#define MPER (MXc*MYc*MZc)
#define VOL  (WXc*WYc*WZc)

// LDS strides (padded: +16B per row so 16 consecutive rows hit distinct banks)
#define XB_STRIDE 136   // bf16 elements per row (128 + 8)
#define WT_STRIDE 136
#define H_STRIDE  132   // f32 elements per row (128 + 4)

// gfx1250 async global->LDS path (guarded so the file always compiles)
#if defined(__gfx1250__) && __has_builtin(__builtin_amdgcn_global_load_async_to_lds_b128)
#define HAVE_ASYNC_LDS 1
// The builtin takes int4-vector pointers (per hipcc diagnostic):
typedef __attribute__((__vector_size__(4 * sizeof(int)))) int b128_t;
typedef __attribute__((address_space(1))) b128_t* gptr_b128;
typedef __attribute__((address_space(3))) b128_t* lptr_b128;
#endif

// ---------------------------------------------------------------- keys ------
__global__ __launch_bounds__(256) void k_keys(const int* __restrict__ coords,
    unsigned* __restrict__ kx, unsigned* __restrict__ ky,
    unsigned* __restrict__ idx, int n)
{
  int i = blockIdx.x * 256 + threadIdx.x;
  if (i >= n) return;
  long long b = coords[i*4+0], z = coords[i*4+1], y = coords[i*4+2], x = coords[i*4+3];
  long long wcx = x / WXc, wcy = y / WYc, wcz = z / WZc;
  long long cix = x % WXc, ciy = y % WYc, ciz = z % WZc;
  long long bwx = b*MPER + wcx*(MYc*MZc) + wcy*MZc + wcz;
  long long bwy = b*MPER + wcy*(MXc*MZc) + wcx*MZc + wcz;
  long long vx = bwx*VOL + cix*(WYc*WZc) + ciy*WZc + ciz;   // < 2^28
  long long vy = bwy*VOL + ciy*(WXc*WZc) + cix*WZc + ciz;
  kx[i] = (unsigned)vx;
  ky[i] = (unsigned)vy;
  idx[i] = (unsigned)i;
}

__global__ __launch_bounds__(256) void k_init_idx(unsigned* __restrict__ idx, int n)
{
  int i = blockIdx.x * 256 + threadIdx.x;
  if (i < n) idx[i] = (unsigned)i;
}

// ------------------------------------------------------------- radix sort ---
__global__ __launch_bounds__(256) void k_hist(const unsigned* __restrict__ keys,
    unsigned* __restrict__ hist, int shift, int nb, int n)
{
  __shared__ unsigned lh[DIG];
  int t = threadIdx.x;
  if (t < DIG) lh[t] = 0;
  __syncthreads();
  int i = blockIdx.x * 256 + t;
  if (i < n) atomicAdd(&lh[(keys[i] >> shift) & (DIG-1)], 1u);
  __syncthreads();
  if (t < DIG) hist[(size_t)t * nb + blockIdx.x] = lh[t];
}

__global__ __launch_bounds__(DIG) void k_scan(unsigned* __restrict__ hist, int nb)
{
  __shared__ unsigned tot[DIG];
  __shared__ unsigned base[DIG];
  int d = threadIdx.x;
  unsigned s = 0;
  for (int b = 0; b < nb; ++b) s += hist[(size_t)d * nb + b];
  tot[d] = s;
  __syncthreads();
  if (d == 0) {
    unsigned run = 0;
    for (int i = 0; i < DIG; ++i) { base[i] = run; run += tot[i]; }
  }
  __syncthreads();
  unsigned run = base[d];
  for (int b = 0; b < nb; ++b) {
    unsigned v = hist[(size_t)d * nb + b];
    hist[(size_t)d * nb + b] = run;
    run += v;
  }
}

__global__ __launch_bounds__(256) void k_scatter(const unsigned* __restrict__ keys,
    const unsigned* __restrict__ idx, unsigned* __restrict__ keyso,
    unsigned* __restrict__ idxo, const unsigned* __restrict__ hist,
    int shift, int nb, int n)
{
  __shared__ unsigned sdig[256];
  int t = threadIdx.x;
  int i = blockIdx.x * 256 + t;
  unsigned key = 0, id = 0, d = 0;
  bool valid = (i < n);
  if (valid) { key = keys[i]; id = idx[i]; d = (key >> shift) & (DIG-1); }
  sdig[t] = valid ? d : 0xffffffffu;
  __syncthreads();
  if (valid) {
    unsigned rank = 0;
    for (int j = 0; j < t; ++j) rank += (sdig[j] == d) ? 1u : 0u;  // stable rank
    unsigned pos = hist[(size_t)d * nb + blockIdx.x] + rank;
    keyso[pos] = key;
    idxo[pos] = id;
  }
}

// ----------------------------------------------------------- WMMA helpers ---
__device__ __forceinline__ v16bf mk16(v8bf lo, v8bf hi)
{
  v16bf r;
#pragma unroll
  for (int i = 0; i < 8; ++i) { r[i] = lo[i]; r[i+8] = hi[i]; }
  return r;
}

// A fragment (16x32 bf16): lane l<16 -> row m0+l, K {kb*32+0..7, +16..23};
// lanes 16..31 -> row m0+l-16, K {+8..15, +24..31}
__device__ __forceinline__ v16bf load_a(const __bf16* xb, int m0, int kb, int l, int hf)
{
  const __bf16* p = xb + (m0 + l) * XB_STRIDE + kb * 32 + hf * 8;
  return mk16(*(const v8bf*)p, *(const v8bf*)(p + 16));
}

// B fragment (32x16 bf16) from transposed weights wt[n][k]:
// lane l<16 -> col n0+l, K {kb*32+0..7, +16..23} ; lanes 16..31 -> col n0+l-16, K {+8..15, +24..31}
__device__ __forceinline__ v16bf load_b(const __bf16* wt, int n0, int kb, int l, int hf)
{
  const __bf16* p = wt + (n0 + l) * WT_STRIDE + kb * 32 + hf * 8;
  return mk16(*(const v8bf*)p, *(const v8bf*)(p + 16));
}

// One wave computes rows [m0, m0+32) x all 128 cols: 16 frags, K=128 in 4 steps
__device__ __forceinline__ void gemm_wave(const __bf16* xb, const __bf16* wt,
                                          int m0, int l, int hf,
                                          v8f acc0[8], v8f acc1[8])
{
  const v8f z = {0.f,0.f,0.f,0.f,0.f,0.f,0.f,0.f};
#pragma unroll
  for (int j = 0; j < 8; ++j) { acc0[j] = z; acc1[j] = z; }
#pragma unroll
  for (int kb = 0; kb < 4; ++kb) {
    v16bf a0 = load_a(xb, m0,      kb, l, hf);
    v16bf a1 = load_a(xb, m0 + 16, kb, l, hf);
#pragma unroll
    for (int j = 0; j < 8; ++j) {
      v16bf b = load_b(wt, j * 16, kb, l, hf);
      acc0[j] = __builtin_amdgcn_wmma_f32_16x16x32_bf16(false, a0, false, b,
                    (short)0, acc0[j], false, false);
      acc1[j] = __builtin_amdgcn_wmma_f32_16x16x32_bf16(false, a1, false, b,
                    (short)0, acc1[j], false, false);
    }
  }
}

// -------------------------------------------------------- fused block ------
// One workgroup per group of 256 rows. 235 KB LDS (CDNA5: 320 KB / WGP).
__global__ __launch_bounds__(256) void k_block(float* __restrict__ feat,
    const unsigned* __restrict__ perm,
    const float* __restrict__ Win, const float* __restrict__ Wgate,
    const float* __restrict__ Wout, const float* __restrict__ loga)
{
  __shared__ __bf16 xb[GROUP * XB_STRIDE];   // x (then y), bf16
  __shared__ float  hb[GROUP * H_STRIDE];    // async staging -> u -> h (f32)
  __shared__ __bf16 wt[C * WT_STRIDE];       // current W^T, bf16
  __shared__ int    sidx[GROUP];

  const int t    = threadIdx.x;
  const int lane = t & 31;
  const int l    = lane & 15;
  const int hf   = lane >> 4;
  const int m0   = (t >> 5) * 32;            // wave's first row

  __builtin_prefetch(Win, 0, 0);
  __builtin_prefetch(Wgate, 0, 0);
  __builtin_prefetch(Wout, 0, 0);

  // ---- gather rows (blocks own disjoint permutation slices -> race-free)
  {
    int row = (int)perm[blockIdx.x * GROUP + t];
    sidx[t] = row;
    float* gsrc = feat + (size_t)row * C;
#if HAVE_ASYNC_LDS
    // Async DMA the f32 row into hb (dead until GEMM1, wave-private rows),
    // no VGPR round-trip; tracked on ASYNCcnt.
    float* lrow = &hb[t * H_STRIDE];
    {
      gptr_b128 g  = (gptr_b128)gsrc;
      lptr_b128 ld = (lptr_b128)lrow;
#pragma unroll
      for (int i = 0; i < (C * 4) / 16; ++i)        // 32 x b128
        __builtin_amdgcn_global_load_async_to_lds_b128(g + i, ld + i, 0, 0);
    }
#if __has_builtin(__builtin_amdgcn_s_wait_asynccnt)
    __builtin_amdgcn_s_wait_asynccnt(0);
#else
    asm volatile("s_wait_asynccnt 0" ::: "memory");
#endif
    // convert own row f32 -> bf16 with packed 16B LDS stores
#pragma unroll
    for (int i = 0; i < C / 8; ++i) {
      float4 va = *(const float4*)(lrow + i * 8);
      float4 vb = *(const float4*)(lrow + i * 8 + 4);
      v8bf o;
      o[0] = (__bf16)va.x; o[1] = (__bf16)va.y; o[2] = (__bf16)va.z; o[3] = (__bf16)va.w;
      o[4] = (__bf16)vb.x; o[5] = (__bf16)vb.y; o[6] = (__bf16)vb.z; o[7] = (__bf16)vb.w;
      *(v8bf*)&xb[t * XB_STRIDE + i * 8] = o;
    }
#else
    const float4* src = (const float4*)gsrc;
#pragma unroll
    for (int i = 0; i < C / 4; ++i) {
      float4 v = src[i];
      __bf16* d = &xb[t * XB_STRIDE + i * 4];
      d[0] = (__bf16)v.x; d[1] = (__bf16)v.y;
      d[2] = (__bf16)v.z; d[3] = (__bf16)v.w;
    }
#endif
  }
  // ---- stage Win^T
  for (int e = t; e < C * C; e += 256) {
    int k = e >> 7, n = e & 127;
    wt[n * WT_STRIDE + k] = (__bf16)Win[e];
  }
  __syncthreads();

  v8f acc0[8], acc1[8];

  // ---- GEMM1: u = x @ Win  -> hb (f32)
  gemm_wave(xb, wt, m0, l, hf, acc0, acc1);
#pragma unroll
  for (int j = 0; j < 8; ++j)
#pragma unroll
    for (int r = 0; r < 8; ++r) {
      int n = j * 16 + l;
      hb[(m0 +      hf * 8 + r) * H_STRIDE + n] = acc0[j][r];
      hb[(m0 + 16 + hf * 8 + r) * H_STRIDE + n] = acc1[j][r];
    }
  __syncthreads();

  // ---- sequential gated scan over the group dim (channel-parallel, f32)
  if (t < C) {
    float a = __expf(-__expf(loga[t]));
    float h = 0.f;
    for (int g = 0; g < GROUP; ++g) {
      float u = hb[g * H_STRIDE + t];
      h = fmaf(a, h, u);
      hb[g * H_STRIDE + t] = h;
    }
  }
  __syncthreads();

  // ---- stage Wgate^T
  for (int e = t; e < C * C; e += 256) {
    int k = e >> 7, n = e & 127;
    wt[n * WT_STRIDE + k] = (__bf16)Wgate[e];
  }
  __syncthreads();

  // ---- GEMM2: g = x @ Wgate ; y = h * silu(g) -> overwrite xb (wave-private rows)
  gemm_wave(xb, wt, m0, l, hf, acc0, acc1);
#pragma unroll
  for (int j = 0; j < 8; ++j)
#pragma unroll
    for (int r = 0; r < 8; ++r) {
      int n = j * 16 + l;
      {
        int m = m0 + hf * 8 + r;
        float g = acc0[j][r];
        float s = g / (1.f + __expf(-g));
        xb[m * XB_STRIDE + n] = (__bf16)(hb[m * H_STRIDE + n] * s);
      }
      {
        int m = m0 + 16 + hf * 8 + r;
        float g = acc1[j][r];
        float s = g / (1.f + __expf(-g));
        xb[m * XB_STRIDE + n] = (__bf16)(hb[m * H_STRIDE + n] * s);
      }
    }
  __syncthreads();

  // ---- stage Wout^T
  for (int e = t; e < C * C; e += 256) {
    int k = e >> 7, n = e & 127;
    wt[n * WT_STRIDE + k] = (__bf16)Wout[e];
  }
  __syncthreads();

  // ---- GEMM3: out = x + y @ Wout ; scatter back (residual x re-read from L2)
  gemm_wave(xb, wt, m0, l, hf, acc0, acc1);
#pragma unroll
  for (int j = 0; j < 8; ++j)
#pragma unroll
    for (int r = 0; r < 8; ++r) {
      int n = j * 16 + l;
      {
        int m = m0 + hf * 8 + r;
        float* p = feat + (size_t)sidx[m] * C + n;
        *p = *p + acc0[j][r];
      }
      {
        int m = m0 + 16 + hf * 8 + r;
        float* p = feat + (size_t)sidx[m] * C + n;
        *p = *p + acc1[j][r];
      }
    }
}

// ---------------------------------------------------------------- launch ----
extern "C" void kernel_launch(void* const* d_in, const int* in_sizes, int n_in,
                              void* d_out, int out_size, void* d_ws, size_t ws_size,
                              hipStream_t stream)
{
  const float* feats  = (const float*)d_in[0];
  const int*   coords = (const int*)  d_in[1];
  const float* w_in   = (const float*)d_in[2];
  const float* w_gate = (const float*)d_in[3];
  const float* w_out  = (const float*)d_in[4];
  const float* log_a  = (const float*)d_in[5];
  float* feat = (float*)d_out;

  const int n    = in_sizes[0] / C;   // 196608
  const int nb   = (n + 255) / 256;   // 768 (n is a multiple of 256)
  const int ngrp = n / GROUP;         // 768

  // workspace carve (all u32): kx, ky, kt, ia, it, hist
  unsigned* kx   = (unsigned*)d_ws;
  unsigned* ky   = kx + n;
  unsigned* kt   = ky + n;
  unsigned* ia   = kt + n;
  unsigned* it   = ia + n;
  unsigned* hist = it + n;            // DIG * nb entries

  // working copy of features (updated in place by both passes)
  (void)hipMemcpyAsync(feat, feats, (size_t)n * C * sizeof(float),
                       hipMemcpyDeviceToDevice, stream);

  k_keys<<<nb, 256, 0, stream>>>(coords, kx, ky, ia, n);

  for (int pass = 0; pass < 2; ++pass) {
    unsigned* keys = (pass == 0) ? kx : ky;
    if (pass == 1) k_init_idx<<<nb, 256, 0, stream>>>(ia, n);

    unsigned *kc = keys, *ic = ia, *kn = kt, *io = it;
    for (int p = 0; p < 4; ++p) {                 // 4 x 7 bits covers 28-bit keys
      int shift = p * RBITS;
      k_hist   <<<nb, 256, 0, stream>>>(kc, hist, shift, nb, n);
      k_scan   <<<1, DIG, 0, stream>>>(hist, nb);
      k_scatter<<<nb, 256, 0, stream>>>(kc, ic, kn, io, hist, shift, nb, n);
      unsigned* tk = kc; kc = kn; kn = tk;
      unsigned* ti = ic; ic = io; io = ti;
    }
    // after 4 swaps the sorted (keys, perm) are back in (keys, ia)

    k_block<<<ngrp, 256, 0, stream>>>(feat, ia,
        w_in  + (size_t)pass * C * C,
        w_gate+ (size_t)pass * C * C,
        w_out + (size_t)pass * C * C,
        log_a + (size_t)pass * C);
  }
}
